// WindowAttention_72447508349161
// MI455X (gfx1250) — matrix operand order repaired and checked by reading, not compiled, not run on hardware
//
#include <hip/hip_runtime.h>
#include <hip/hip_bf16.h>

// ---------------------------------------------------------------------------
// Swin window attention, fully fused: one window (49 tokens) per workgroup.
// All matmuls on v_wmma_f32_16x16x32_f16; x staged into LDS with the CDNA5
// async-to-LDS path (global_load_async_to_lds_b128 + s_wait_asynccnt).
// ---------------------------------------------------------------------------

typedef __attribute__((ext_vector_type(16))) _Float16 v16h;
typedef __attribute__((ext_vector_type(8)))  _Float16 v8h;
typedef __attribute__((ext_vector_type(8)))  float    v8f;
typedef __attribute__((ext_vector_type(4)))  _Float16 h4;
typedef __attribute__((ext_vector_type(4)))  float    f4;

#define TDIM 384
#define TNH  12
#define THD  32
#define TN   49
#define TNP  64          // tokens padded to 64 (4 M-tiles of 16)
#define TNW  1024
__device__ __constant__ float kScale = 0.17677669529663687f;  // 32^-0.5

union FragU { v16h v; v8h h[2]; };

// A operand (M x 32 K-slice) from row-major f16 source.
// 16-bit A layout: lanes 0-15 hold K[0..7],K[16..23]; lanes 16-31 K[8..15],K[24..31].
__device__ __forceinline__ v16h load_frag_A(const _Float16* __restrict__ src,
                                            int rowStride, int mBase, int kBase, int lane) {
    const int half = lane >> 4;
    const _Float16* p = src + (mBase + (lane & 15)) * rowStride + kBase + half * 8;
    FragU f;
    f.h[0] = *(const v8h*)(p);
    f.h[1] = *(const v8h*)(p + 16);
    return f.v;
}

// B operand (32 K-slice x N): source is row-major [Nrows][K] where each row is a
// column of B. Lanes 0-15 hold K[0..7],K[16..23]; lanes 16-31 K[8..15],K[24..31] of column (lane&15).
__device__ __forceinline__ v16h load_frag_B(const _Float16* __restrict__ src,
                                            int rowStride, int nBase, int kBase, int lane) {
    const _Float16* p = src + (nBase + (lane & 15)) * rowStride + kBase + (lane >> 4) * 8;
    FragU f;
    f.h[0] = *(const v8h*)(p);
    f.h[1] = *(const v8h*)(p + 16);
    return f.v;
}

__device__ __forceinline__ v8f wmma32(v16h a, v16h b, v8f c) {
    // (neg_a, A, neg_b, B, c_mod, C, reuse_a, reuse_b)
    return __builtin_amdgcn_wmma_f32_16x16x32_f16(false, a, false, b, (short)0, c, false, false);
}

// ---------------------------------------------------------------------------
// Prep: fp32 weights -> f16 in workspace; gather relative-position bias to
// rbias[h][i][j] (f32).
// ---------------------------------------------------------------------------
__global__ void swin_prep_kernel(const float* __restrict__ qkv_w,
                                 const float* __restrict__ proj_w,
                                 const float* __restrict__ rpb,
                                 const int*   __restrict__ rel_idx,
                                 _Float16* __restrict__ qkvw_h,
                                 _Float16* __restrict__ projw_h,
                                 float*    __restrict__ rbias) {
    const int i = blockIdx.x * blockDim.x + threadIdx.x;
    const int stride = gridDim.x * blockDim.x;
    for (int t = i; t < 3 * TDIM * TDIM; t += stride) qkvw_h[t] = (_Float16)qkv_w[t];
    for (int t = i; t < TDIM * TDIM;     t += stride) projw_h[t] = (_Float16)proj_w[t];
    for (int t = i; t < TNH * TN * TN;   t += stride) {
        const int h = t / (TN * TN);
        const int ij = t % (TN * TN);
        rbias[t] = rpb[rel_idx[ij] * TNH + h];
    }
}

// ---------------------------------------------------------------------------
// Fused window-attention kernel. Block = 256 threads = 8 wave32.
// LDS (240 KB total, carved from one buffer so aliasing is explicit):
//   [0      , 49152 ) xs    : x as f16 [64][384], reused as attention output
//   [49152  , 98304 ) qs    : q*scale  [12][64][32] f16
//   [98304  , 147456) ksm   : k        [12][64][32] f16
//   [147456 , 196608) vt    : v^T      [12][32][64] f16
//   [196608 , 229376) sc    : scores   [2][64][64]  f32
//   [229376 , 245760) at    : probs    [2][64][64]  f16
//   [49152  , 124416) stage : x fp32 staging (overlays qs/ksm; dead by phase 1)
// ---------------------------------------------------------------------------
__global__ __launch_bounds__(256) void swin_attn_kernel(
    const float*    __restrict__ x,
    const float*    __restrict__ mask,
    const float*    __restrict__ qkv_b,
    const float*    __restrict__ proj_b,
    const _Float16* __restrict__ qkvw,
    const _Float16* __restrict__ projw,
    const float*    __restrict__ rbias,
    float*          __restrict__ out) {

    __shared__ __align__(16) char smem[245760];
    _Float16* xs    = (_Float16*)(smem);
    _Float16* qs    = (_Float16*)(smem + 49152);
    _Float16* ksm   = (_Float16*)(smem + 98304);
    _Float16* vt    = (_Float16*)(smem + 147456);
    float*    sc    = (float*)   (smem + 196608);
    _Float16* at    = (_Float16*)(smem + 229376);
    float*    stage = (float*)   (smem + 49152);   // overlays qs/ksm during phase 0

    const int win  = blockIdx.x;
    const int tid  = threadIdx.x;
    const int wave = tid >> 5;
    const int lane = tid & 31;

    // ---- Phase 0a: async DMA of the x window (fp32) into LDS staging ----
    const float* __restrict__ xw = x + (size_t)win * TN * TDIM;
    for (int t = tid; t < TN * TDIM / 4; t += 256) {        // 4704 x b128
        const float* g = xw + t * 4;
        const unsigned l = (unsigned)(size_t)(stage + t * 4);
        asm volatile("global_load_async_to_lds_b128 %0, %1, off"
                     :: "v"(l), "v"(g) : "memory");
    }
    asm volatile("s_wait_asynccnt 0x0" ::: "memory");
    __syncthreads();

    // ---- Phase 0b: convert staged fp32 -> f16, zero-pad rows 49..63 ----
    for (int t = tid; t < TNP * (TDIM / 4); t += 256) {
        const int m = t / (TDIM / 4), kc = t % (TDIM / 4);
        f4 v = {0.f, 0.f, 0.f, 0.f};
        if (m < TN) v = *(const f4*)(stage + m * TDIM + kc * 4);
        h4 hv = {(_Float16)v[0], (_Float16)v[1], (_Float16)v[2], (_Float16)v[3]};
        *(h4*)(&xs[m * TDIM + kc * 4]) = hv;
    }
    __syncthreads();

    // ---- Phase 1: QKV GEMM (64x1152 = 4 x 72 tiles, K=384) ----
    for (int t = wave; t < 4 * 72; t += 8) {
        const int mt = t & 3, nt = t >> 2;
        v8f acc = {};
        for (int kk = 0; kk < 12; ++kk) {
            if (kk + 1 < 12)
                __builtin_prefetch(qkvw + (nt * 16 + (lane & 15)) * TDIM + (kk + 1) * 32, 0, 3);
            v16h a = load_frag_A(xs,   TDIM, mt * 16, kk * 32, lane);
            v16h b = load_frag_B(qkvw, TDIM, nt * 16, kk * 32, lane);
            acc = wmma32(a, b, acc);
        }
        const int nAbs = nt * 16 + (lane & 15);
        const float bias = qkv_b[nAbs];
        const int kind = nAbs / TDIM;        // 0=q, 1=k, 2=v
        const int ch = nAbs % TDIM;
        const int h = ch / THD, d = ch % THD;
#pragma unroll
        for (int r = 0; r < 8; ++r) {
            const int m = mt * 16 + r + 8 * (lane >> 4);
            const float vv = acc[r] + bias;
            if (kind == 0)      qs [(h * TNP + m) * THD + d] = (_Float16)(vv * kScale);
            else if (kind == 1) ksm[(h * TNP + m) * THD + d] = (_Float16)vv;
            else                vt [(h * THD + d) * TNP + m] = (_Float16)vv;
        }
    }
    __syncthreads();

    // ---- Phase 2: attention, two heads per pass ----
    const float* __restrict__ maskw = mask + (size_t)(win % TNW) * TN * TN;
    for (int hc = 0; hc < TNH; hc += 2) {
        // scores = q @ k^T (+bias +mask); 2 heads x 16 tiles, K=32 (one WMMA)
        for (int t = wave; t < 32; t += 8) {
            const int hl = t >> 4, tl = t & 15, mt = tl & 3, jt = tl >> 2;
            const int h = hc + hl;
            v16h a = load_frag_A(qs  + h * TNP * THD, THD, mt * 16, 0, lane);
            v16h b = load_frag_B(ksm + h * TNP * THD, THD, jt * 16, 0, lane);
            v8f acc = {};
            acc = wmma32(a, b, acc);
            const int j = jt * 16 + (lane & 15);
#pragma unroll
            for (int r = 0; r < 8; ++r) {
                const int m = mt * 16 + r + 8 * (lane >> 4);
                float s = -1e30f;
                if (m < TN && j < TN)
                    s = acc[r] + rbias[(h * TN + m) * TN + j] + maskw[m * TN + j];
                sc[(hl * TNP + m) * TNP + j] = s;
            }
        }
        __syncthreads();

        // softmax, one row per thread (2*64 rows)
        if (tid < 2 * TNP) {
            const int hl = tid >> 6, m = tid & 63;
            _Float16* __restrict__ arow = at + (hl * TNP + m) * TNP;
            if (m < TN) {
                const float* __restrict__ srow = sc + (hl * TNP + m) * TNP;
                float mx = -1e30f;
                for (int j = 0; j < TN; ++j) mx = fmaxf(mx, srow[j]);
                float sum = 0.f;
                for (int j = 0; j < TN; ++j) sum += __expf(srow[j] - mx);
                const float inv = 1.0f / sum;
                for (int j = 0; j < TNP; ++j)
                    arow[j] = (_Float16)((j < TN) ? __expf(srow[j] - mx) * inv : 0.f);
            } else {
                for (int j = 0; j < TNP; ++j) arow[j] = (_Float16)0.f;
            }
        }
        __syncthreads();

        // out = attn @ v ; 2 heads x (4 M-tiles x 2 D-tiles), K=64 (2 WMMA)
        for (int t = wave; t < 16; t += 8) {
            const int hl = t >> 3, rem = t & 7, mt = rem & 3, dt = rem >> 2;
            const int h = hc + hl;
            v8f acc = {};
            for (int kk = 0; kk < 2; ++kk) {
                v16h a = load_frag_A(at + hl * TNP * TNP, TNP, mt * 16, kk * 32, lane);
                v16h b = load_frag_B(vt + h * THD * TNP,  TNP, dt * 16, kk * 32, lane);
                acc = wmma32(a, b, acc);
            }
            const int d = dt * 16 + (lane & 15);
#pragma unroll
            for (int r = 0; r < 8; ++r) {
                const int m = mt * 16 + r + 8 * (lane >> 4);
                xs[m * TDIM + h * THD + d] = (_Float16)acc[r];   // xs reused as attn-out
            }
        }
        __syncthreads();
    }

    // ---- Phase 3: projection GEMM (64x384, K=384), f32 output ----
    float* __restrict__ outw = out + (size_t)win * TN * TDIM;
    for (int t = wave; t < 4 * 24; t += 8) {
        const int mt = t & 3, nt = t >> 2;
        v8f acc = {};
        for (int kk = 0; kk < 12; ++kk) {
            if (kk + 1 < 12)
                __builtin_prefetch(projw + (nt * 16 + (lane & 15)) * TDIM + (kk + 1) * 32, 0, 3);
            v16h a = load_frag_A(xs,    TDIM, mt * 16, kk * 32, lane);
            v16h b = load_frag_B(projw, TDIM, nt * 16, kk * 32, lane);
            acc = wmma32(a, b, acc);
        }
        const int nAbs = nt * 16 + (lane & 15);
        const float bias = proj_b[nAbs];
#pragma unroll
        for (int r = 0; r < 8; ++r) {
            const int m = mt * 16 + r + 8 * (lane >> 4);
            if (m < TN) outw[m * TDIM + nAbs] = acc[r] + bias;
        }
    }
}

// ---------------------------------------------------------------------------
extern "C" void kernel_launch(void* const* d_in, const int* in_sizes, int n_in,
                              void* d_out, int out_size, void* d_ws, size_t ws_size,
                              hipStream_t stream) {
    const float* x      = (const float*)d_in[0];
    const float* mask   = (const float*)d_in[1];
    const float* rpb    = (const float*)d_in[2];
    const float* qkv_w  = (const float*)d_in[3];
    const float* qkv_b  = (const float*)d_in[4];
    const float* proj_w = (const float*)d_in[5];
    const float* proj_b = (const float*)d_in[6];
    const int*   relidx = (const int*)d_in[7];

    char* ws = (char*)d_ws;
    _Float16* qkvw_h  = (_Float16*)ws;                      // 1152*384 f16 = 884,736 B
    _Float16* projw_h = (_Float16*)(ws + 884736);           // 384*384 f16  = 294,912 B
    float*    rbias   = (float*)(ws + 884736 + 294912);     // 12*49*49 f32 = 115,248 B

    swin_prep_kernel<<<512, 256, 0, stream>>>(qkv_w, proj_w, rpb, relidx,
                                              qkvw_h, projw_h, rbias);
    swin_attn_kernel<<<4096, 256, 0, stream>>>(x, mask, qkv_b, proj_b,
                                               qkvw_h, projw_h, rbias, (float*)d_out);
}
